// DGCNN_cls_14336600834838
// MI455X (gfx1250) — matrix operand order repaired and checked by reading, not compiled, not run on hardware
//
#include <hip/hip_runtime.h>

#define BB 16
#define NN 1024
#define KNN_K 20
#define NEG_SLOPE 0.2f

typedef __attribute__((ext_vector_type(16))) __bf16 v16bf;
typedef __attribute__((ext_vector_type(8)))  float  v8f;

// ---- WMMA 16-bit A-matrix K index for VGPR pair v (ISA 7.12.2) ----
// lanes 0-15: v0..3 -> K=0..7, v4..7 -> K=16..23 ; lanes 16-31: +8 / +8
__device__ __forceinline__ int a_kmap(int v, int lane) {
  int kb = (lane & 16) ? 8 : 0;
  return kb + ((v < 4) ? (2 * v) : (16 + 2 * (v - 4)));
}
// B-matrix: lane = N column (0-15 twice), same K order as A: v0..3 -> K=0..7, v4..7 -> K=16..23; lanes 16-31 +8
__device__ __forceinline__ int b_kmap(int v, int lane) {
  return ((lane & 16) ? 8 : 0) + ((v < 4) ? (2 * v) : (16 + 2 * (v - 4)));
}
__device__ __forceinline__ float lrelu(float v) { return v >= 0.f ? v : NEG_SLOPE * v; }

// ---------------- squared norms ----------------
__global__ void sqnorm_kernel(const float* __restrict__ feat, float* __restrict__ sq, int C) {
  int i = blockIdx.x * blockDim.x + threadIdx.x;
  if (i >= BB * NN) return;
  const float* p = feat + (size_t)i * C;
  float s = 0.f;
  for (int c = 0; c < C; ++c) s += p[c] * p[c];
  sq[i] = s;
}

// ---------------- brute-force KNN top-20 ----------------
// rank key: 2*dot(x_n,x_m) - |x_m|^2  (dropping the per-row constant -|x_n|^2)
// C is compile-time so the query vector stays resident in VGPRs across the m-loop.
template <int C>
__global__ void knn_kernel(const float* __restrict__ feat, const float* __restrict__ sq,
                           int* __restrict__ idx) {
  int i = blockIdx.x * blockDim.x + threadIdx.x;
  if (i >= BB * NN) return;
  int b = i / NN, n = i % NN;
  const float* xb  = feat + (size_t)b * NN * C;
  const float* xn  = xb + (size_t)n * C;
  const float* sqb = sq + (size_t)b * NN;

  float q[C];
  #pragma unroll
  for (int c = 0; c < C; ++c) q[c] = xn[c];

  float bd[KNN_K]; int bi[KNN_K];
  for (int j = 0; j < KNN_K; ++j) { bd[j] = -3.0e38f; bi[j] = 0; }
  for (int m = 0; m < NN; ++m) {
    const float* xm = xb + (size_t)m * C;
    float dot = 0.f;
    #pragma unroll
    for (int c = 0; c < C; ++c) dot += q[c] * xm[c];
    float d = 2.f * dot - sqb[m];
    if (d > bd[KNN_K - 1]) {
      int j = KNN_K - 1;
      while (j > 0 && bd[j - 1] < d) { bd[j] = bd[j - 1]; bi[j] = bi[j - 1]; --j; }
      bd[j] = d; bi[j] = m;
    }
  }
  int* op = idx + (size_t)i * KNN_K;
  for (int j = 0; j < KNN_K; ++j) op[j] = bi[j];
}

// ---------------- fused edge-conv (gather + WMMA GEMM + BN + lrelu + max over k) ----------------
template <int CIN, int COUT>
__global__ void __launch_bounds__(32)
edgeconv_kernel(const float* __restrict__ feat, const int* __restrict__ idx,
                const float* __restrict__ W, const float* __restrict__ g,
                const float* __restrict__ bsh, float* __restrict__ out) {
  constexpr int NT   = 4;               // points per workgroup
  constexpr int ROWS = NT * KNN_K;      // 80 edge rows = 5 tiles of 16
  constexpr int KDIM = 2 * CIN;
  constexpr int KPAD = (KDIM < 32) ? 32 : KDIM;
  __shared__ __bf16 e_lds[ROWS * KPAD];
  __shared__ float  h_lds[ROWS * 16];

  const int lane = threadIdx.x;
  const int b    = blockIdx.x / (NN / NT);
  const int n0   = (blockIdx.x % (NN / NT)) * NT;

  // gather edge features [nb-ctr, ctr] into LDS (bf16), zero-pad K to KPAD
  for (int r = 0; r < ROWS; ++r) {
    int nl = r / KNN_K, kk = r % KNN_K;
    int n  = n0 + nl;
    int j  = idx[((size_t)b * NN + n) * KNN_K + kk];
    const float* xc = feat + ((size_t)b * NN + n) * CIN;
    const float* xj = feat + ((size_t)b * NN + j) * CIN;
    for (int c = lane; c < CIN; c += 32) {
      float cv = xc[c];
      e_lds[r * KPAD + c]       = (__bf16)(xj[c] - cv);
      e_lds[r * KPAD + CIN + c] = (__bf16)cv;
    }
    for (int c = KDIM + lane; c < KPAD; c += 32) e_lds[r * KPAD + c] = (__bf16)0.f;
  }
  __syncthreads();

  const float bnsc = rsqrtf(1.f + 1e-5f);

  for (int ot = 0; ot < COUT / 16; ++ot) {
    const int o0 = ot * 16;
    const int ch = o0 + (lane & 15);       // each lane owns one output channel column
    const float scale = g[ch] * bnsc;
    const float shift = bsh[ch];
    const float* wr = W + (size_t)ch * KDIM;
    __builtin_prefetch(wr, 0, 1);          // global_prefetch_b8: pull weight row toward L0/L2

    for (int t = 0; t < ROWS / 16; ++t) {
      v8f acc = {};
      const __bf16* ar = e_lds + (size_t)(t * 16 + (lane & 15)) * KPAD;
      #pragma unroll
      for (int kc = 0; kc < KPAD; kc += 32) {
        v16bf A, Bv;
        #pragma unroll
        for (int v = 0; v < 8; ++v) {
          int ka = kc + a_kmap(v, lane);
          A[2 * v]     = ar[ka];
          A[2 * v + 1] = ar[ka + 1];
          int kb = kc + b_kmap(v, lane);
          float w0 = (kb < KDIM)     ? wr[kb]     : 0.f;
          float w1 = (kb + 1 < KDIM) ? wr[kb + 1] : 0.f;
          Bv[2 * v]     = (__bf16)w0;
          Bv[2 * v + 1] = (__bf16)w1;
        }
        acc = __builtin_amdgcn_wmma_f32_16x16x32_bf16(false, A, false, Bv, (short)0, acc,
                                                      false, false);
      }
      const int rbase = t * 16 + ((lane & 16) ? 8 : 0);   // D: VGPR r -> row, lane half splits M
      #pragma unroll
      for (int r = 0; r < 8; ++r)
        h_lds[(rbase + r) * 16 + (lane & 15)] = lrelu(acc[r] * scale + shift);
    }
    __syncthreads();
    // max over k=20 per point / channel
    for (int e = lane; e < NT * 16; e += 32) {
      int nl = e >> 4, c = e & 15;
      float m = h_lds[(nl * KNN_K) * 16 + c];
      for (int kk = 1; kk < KNN_K; ++kk)
        m = fmaxf(m, h_lds[(nl * KNN_K + kk) * 16 + c]);
      out[((size_t)b * NN + n0 + nl) * COUT + o0 + c] = m;
    }
    __syncthreads();
  }
}

// ---------------- pointwise 512 -> 1024 GEMM over concat(h1..h4) ----------------
__global__ void __launch_bounds__(32)
linear5_kernel(const float* __restrict__ h1, const float* __restrict__ h2,
               const float* __restrict__ h3, const float* __restrict__ h4,
               const float* __restrict__ W5, const float* __restrict__ g5,
               const float* __restrict__ b5, float* __restrict__ h5) {
  const int lane = threadIdx.x;
  const int row0 = blockIdx.x * 16;          // flattened (b,n) rows
  const int o0   = blockIdx.y * 16;
  const int arow = row0 + (lane & 15);
  const int ocol = o0 + (lane & 15);
  const float* wr = W5 + (size_t)ocol * 512;
  v8f acc = {};
  for (int k0 = 0; k0 < 512; k0 += 32) {     // chunk stays inside one concat segment
    const float* seg; int cw, off;
    if (k0 < 64)       { seg = h1; cw = 64;  off = k0; }
    else if (k0 < 128) { seg = h2; cw = 64;  off = k0 - 64; }
    else if (k0 < 256) { seg = h3; cw = 128; off = k0 - 128; }
    else               { seg = h4; cw = 256; off = k0 - 256; }
    const float* ap = seg + (size_t)arow * cw + off;
    __builtin_prefetch(wr + k0 + 32, 0, 1);
    v16bf A, Bv;
    #pragma unroll
    for (int v = 0; v < 8; ++v) {
      int ka = a_kmap(v, lane);
      A[2 * v]     = (__bf16)ap[ka];
      A[2 * v + 1] = (__bf16)ap[ka + 1];
      int kb = k0 + b_kmap(v, lane);
      Bv[2 * v]     = (__bf16)wr[kb];
      Bv[2 * v + 1] = (__bf16)wr[kb + 1];
    }
    acc = __builtin_amdgcn_wmma_f32_16x16x32_bf16(false, A, false, Bv, (short)0, acc,
                                                  false, false);
  }
  const float scale = g5[ocol] * rsqrtf(1.f + 1e-5f);
  const float shift = b5[ocol];
  const int rbase = (lane & 16) ? 8 : 0;
  #pragma unroll
  for (int r = 0; r < 8; ++r)
    h5[(size_t)(row0 + rbase + r) * 1024 + ocol] = lrelu(acc[r] * scale + shift);
}

// ---------------- max/mean pool over N ----------------
__global__ void pool_kernel(const float* __restrict__ h5, float* __restrict__ pooled) {
  int i = blockIdx.x * blockDim.x + threadIdx.x;   // b*1024 + e
  if (i >= BB * 1024) return;
  int b = i >> 10, e = i & 1023;
  const float* p = h5 + (size_t)b * 1024 * 1024 + e;
  float mx = -3.0e38f, sm = 0.f;
  for (int n = 0; n < NN; ++n) {
    float v = p[(size_t)n * 1024];
    mx = fmaxf(mx, v);
    sm += v;
  }
  pooled[b * 2048 + e]        = mx;
  pooled[b * 2048 + 1024 + e] = sm * (1.f / 1024.f);
}

// ---------------- FC layers: 16 rows == exactly one WMMA M-tile ----------------
template <bool HAS_BIAS, bool BN_ACT>
__global__ void __launch_bounds__(32)
fc_kernel(const float* __restrict__ in, const float* __restrict__ W,
          const float* __restrict__ bias, const float* __restrict__ g,
          const float* __restrict__ bsh, float* __restrict__ out, int KD, int O) {
  const int lane = threadIdx.x;
  const int o0   = blockIdx.x * 16;
  const int ocol = o0 + (lane & 15);
  const int ocl  = ocol < O ? ocol : O - 1;   // clamp for masked tail tile
  const float* ap = in + (size_t)(lane & 15) * KD;   // row = batch index
  const float* wr = W + (size_t)ocl * KD;
  v8f acc = {};
  for (int k0 = 0; k0 < KD; k0 += 32) {
    v16bf A, Bv;
    #pragma unroll
    for (int v = 0; v < 8; ++v) {
      int ka = k0 + a_kmap(v, lane);
      A[2 * v]     = (__bf16)ap[ka];
      A[2 * v + 1] = (__bf16)ap[ka + 1];
      int kb = k0 + b_kmap(v, lane);
      Bv[2 * v]     = (__bf16)wr[kb];
      Bv[2 * v + 1] = (__bf16)wr[kb + 1];
    }
    acc = __builtin_amdgcn_wmma_f32_16x16x32_bf16(false, A, false, Bv, (short)0, acc,
                                                  false, false);
  }
  float bia = HAS_BIAS ? bias[ocl] : 0.f;
  float scale = 1.f, shift = 0.f;
  if (BN_ACT) { scale = g[ocl] * rsqrtf(1.f + 1e-5f); shift = bsh[ocl]; }
  const int rbase = (lane & 16) ? 8 : 0;
  #pragma unroll
  for (int r = 0; r < 8; ++r) {
    float v = acc[r] + bia;
    if (BN_ACT) v = lrelu(v * scale + shift);
    if (ocol < O) out[(size_t)(rbase + r) * O + ocol] = v;
  }
}

extern "C" void kernel_launch(void* const* d_in, const int* in_sizes, int n_in,
                              void* d_out, int out_size, void* d_ws, size_t ws_size,
                              hipStream_t stream) {
  (void)in_sizes; (void)n_in; (void)out_size; (void)ws_size;
  const float* x   = (const float*)d_in[0];
  const float* W1  = (const float*)d_in[1];
  const float* g1  = (const float*)d_in[2];
  const float* b1  = (const float*)d_in[3];
  const float* W2  = (const float*)d_in[4];
  const float* g2  = (const float*)d_in[5];
  const float* b2  = (const float*)d_in[6];
  const float* W3  = (const float*)d_in[7];
  const float* g3  = (const float*)d_in[8];
  const float* b3  = (const float*)d_in[9];
  const float* W4  = (const float*)d_in[10];
  const float* g4  = (const float*)d_in[11];
  const float* b4  = (const float*)d_in[12];
  const float* W5  = (const float*)d_in[13];
  const float* g5  = (const float*)d_in[14];
  const float* b5  = (const float*)d_in[15];
  const float* L1  = (const float*)d_in[16];
  const float* g6  = (const float*)d_in[17];
  const float* b6  = (const float*)d_in[18];
  const float* L2  = (const float*)d_in[19];
  const float* l2b = (const float*)d_in[20];
  const float* g7  = (const float*)d_in[21];
  const float* b7  = (const float*)d_in[22];
  const float* L3  = (const float*)d_in[23];
  const float* l3b = (const float*)d_in[24];

  char* ws = (char*)d_ws;
  size_t off = 0;
  auto take = [&](size_t bytes) {
    char* p = ws + off;
    off = (off + bytes + 255) & ~(size_t)255;
    return p;
  };
  float* sq     = (float*)take((size_t)BB * NN * 4);
  int*   idx    = (int*)take((size_t)BB * NN * KNN_K * 4);
  float* h1     = (float*)take((size_t)BB * NN * 64 * 4);
  float* h2     = (float*)take((size_t)BB * NN * 64 * 4);
  float* h3     = (float*)take((size_t)BB * NN * 128 * 4);
  float* h4     = (float*)take((size_t)BB * NN * 256 * 4);
  float* h5     = (float*)take((size_t)BB * NN * 1024 * 4);
  float* pooled = (float*)take((size_t)BB * 2048 * 4);
  float* z1     = (float*)take((size_t)BB * 512 * 4);
  float* z2     = (float*)take((size_t)BB * 256 * 4);

  const int nWG = BB * NN / 4;   // edgeconv workgroups

  // layer 1: x(3) -> h1(64)
  sqnorm_kernel<<<64, 256, 0, stream>>>(x, sq, 3);
  knn_kernel<3><<<64, 256, 0, stream>>>(x, sq, idx);
  edgeconv_kernel<3, 64><<<nWG, 32, 0, stream>>>(x, idx, W1, g1, b1, h1);
  // layer 2: h1(64) -> h2(64)
  sqnorm_kernel<<<64, 256, 0, stream>>>(h1, sq, 64);
  knn_kernel<64><<<64, 256, 0, stream>>>(h1, sq, idx);
  edgeconv_kernel<64, 64><<<nWG, 32, 0, stream>>>(h1, idx, W2, g2, b2, h2);
  // layer 3: h2(64) -> h3(128)
  sqnorm_kernel<<<64, 256, 0, stream>>>(h2, sq, 64);
  knn_kernel<64><<<64, 256, 0, stream>>>(h2, sq, idx);
  edgeconv_kernel<64, 128><<<nWG, 32, 0, stream>>>(h2, idx, W3, g3, b3, h3);
  // layer 4: h3(128) -> h4(256)
  sqnorm_kernel<<<64, 256, 0, stream>>>(h3, sq, 128);
  knn_kernel<128><<<64, 256, 0, stream>>>(h3, sq, idx);
  edgeconv_kernel<128, 256><<<nWG, 32, 0, stream>>>(h3, idx, W4, g4, b4, h4);
  // W5 einsum over concat -> h5(1024)
  linear5_kernel<<<dim3(BB * NN / 16, 1024 / 16), 32, 0, stream>>>(h1, h2, h3, h4, W5, g5, b5, h5);
  // pooling
  pool_kernel<<<64, 256, 0, stream>>>(h5, pooled);
  // FC head
  fc_kernel<false, true><<<512 / 16, 32, 0, stream>>>(pooled, L1, nullptr, g6, b6, z1, 2048, 512);
  fc_kernel<true, true><<<256 / 16, 32, 0, stream>>>(z1, L2, l2b, g7, b7, z2, 512, 256);
  fc_kernel<true, false><<<3, 32, 0, stream>>>(z2, L3, l3b, nullptr, nullptr, (float*)d_out, 256, 40);
}
